// JK_85787676771076
// MI455X (gfx1250) — matrix operand order repaired and checked by reading, not compiled, not run on hardware
//
#include <hip/hip_runtime.h>

typedef __attribute__((ext_vector_type(16))) __bf16 v16bf;
typedef __attribute__((ext_vector_type(8)))  float  v8f;

#define FDIM 128
#define LINDIM 256
#define OUTDIM 10
#define NGRAPH 64

// ---------------- helpers ----------------

__device__ __forceinline__ unsigned short bf16b(float f) {
    // round-to-nearest-even fp32 -> bf16 bits
    unsigned u = __float_as_uint(f);
    unsigned r = u + 0x7FFFu + ((u >> 16) & 1u);
    return (unsigned short)(r >> 16);
}

__device__ __forceinline__ float bf16f(unsigned short h) {
    return __uint_as_float((unsigned)h << 16);   // exact
}

__device__ __forceinline__ void atomicMaxF(float* addr, float val) {
    // monotone-in-float-order bit tricks; valid mixed on same address,
    // works with -inf (0xFF800000) init.
    if (!(val < 0.0f)) atomicMax((int*)addr, __float_as_int(val));
    else               atomicMin((unsigned int*)addr, __float_as_uint(val));
}

// CDNA5 async copy global->LDS (ASYNCcnt path, CDNA5 ISA).
// lds_off = low 32 bits of generic shared pointer (LDS aperture truncation).
__device__ __forceinline__ void async_copy_b128(unsigned lds_off, const void* gaddr) {
    asm volatile("global_load_async_to_lds_b128 %0, %1, off"
                 :: "v"(lds_off), "v"(gaddr) : "memory");
}
__device__ __forceinline__ void wait_async0() {
    asm volatile("s_wait_asynccnt 0" ::: "memory");
}

union Frag16 { uint4 q[2]; v16bf v; };

// ---------------- kernels ----------------

__global__ void fill_u32_kernel(unsigned* __restrict__ p, unsigned v, long n) {
    long i = (long)blockIdx.x * blockDim.x + threadIdx.x;
    if (i < n) p[i] = v;
}

__global__ void fix_neginf_kernel(float* __restrict__ p, long n) {
    long i = (long)blockIdx.x * blockDim.x + threadIdx.x;
    if (i < n && __float_as_uint(p[i]) == 0xFF800000u) p[i] = 0.0f;
}

// fp32 -> bf16, 2 elements per thread (packed u32 store)
__global__ void cvt_f32_bf16_kernel(const float* __restrict__ in,
                                    unsigned short* __restrict__ out, long n2) {
    long i = (long)blockIdx.x * blockDim.x + threadIdx.x;
    if (i >= n2) return;
    float2 f = *(const float2*)&in[i * 2];
    ((unsigned*)out)[i] = (unsigned)bf16b(f.x) | ((unsigned)bf16b(f.y) << 16);
}

// W [128k][128n] fp32 -> WT [128n][128k] bf16 (one-time; makes staging contiguous)
__global__ void wtrans_kernel(const float* __restrict__ W,
                              unsigned short* __restrict__ WT) {
    int i = blockIdx.x * blockDim.x + threadIdx.x;
    if (i >= FDIM * FDIM) return;
    int n = i >> 7, k = i & 127;
    WT[n * FDIM + k] = bf16b(W[k * FDIM + n]);
}

// one thread per (edge, 8-feature chunk): gather bf16 x[src] row (16B),
// scatter-max into fp32 agg[dst]
__global__ void scatter_max_kernel(const unsigned short* __restrict__ xb,
                                   const long long* __restrict__ src,
                                   const long long* __restrict__ dst,
                                   float* __restrict__ agg, long E) {
    long i = (long)blockIdx.x * blockDim.x + threadIdx.x;
    long e = i >> 4;
    int  c = (int)(i & 15) * 8;
    if (e >= E) return;
    long s = (long)src[e];
    long d = (long)dst[e];
    uint4 v = *(const uint4*)&xb[s * FDIM + c];
    float* a = &agg[d * FDIM + c];
    atomicMaxF(a + 0, bf16f((unsigned short)(v.x & 0xFFFF)));
    atomicMaxF(a + 1, bf16f((unsigned short)(v.x >> 16)));
    atomicMaxF(a + 2, bf16f((unsigned short)(v.y & 0xFFFF)));
    atomicMaxF(a + 3, bf16f((unsigned short)(v.y >> 16)));
    atomicMaxF(a + 4, bf16f((unsigned short)(v.z & 0xFFFF)));
    atomicMaxF(a + 5, bf16f((unsigned short)(v.z >> 16)));
    atomicMaxF(a + 6, bf16f((unsigned short)(v.w & 0xFFFF)));
    atomicMaxF(a + 7, bf16f((unsigned short)(v.w >> 16)));
}

// xout(bf16) = relu(agg @ Wn + xin @ Ws + bias); agg fp32, xin/WnT/WsT bf16
__global__ __launch_bounds__(256)
void gnn_layer_kernel(const float* __restrict__ agg,
                      const unsigned short* __restrict__ xin,   // [N][128] bf16
                      const unsigned short* __restrict__ WnT,   // [n][k] bf16
                      const unsigned short* __restrict__ WsT,   // [n][k] bf16
                      const float* __restrict__ bias,           // [128]
                      unsigned short* __restrict__ xout, int N)
{
    __shared__ unsigned short sAg[128][40];   // A tile (agg), [row][k], pad 8
    __shared__ unsigned short sXx[128][40];   // A tile (x)
    __shared__ unsigned short sWn[128][40];   // [n][k]
    __shared__ unsigned short sWs[128][40];   // [n][k]

    const int t    = threadIdx.x;
    const int lane = t & 31;
    const int wv   = t >> 5;               // wave 0..7 -> rows wv*16..wv*16+15
    const int row0 = blockIdx.x * 128;

    const unsigned sX_base  = (unsigned)(unsigned long long)&sXx[0][0];
    const unsigned sWn_base = (unsigned)(unsigned long long)&sWn[0][0];
    const unsigned sWs_base = (unsigned)(unsigned long long)&sWs[0][0];

    v8f acc[8] = {};                       // 8 n-tiles of 16x16 f32

    for (int kc = 0; kc < 4; ++kc) {       // K chunks of 32
        __syncthreads();
        // --- async stage x tile + both weight tiles (pure contiguous copies) ---
        #pragma unroll
        for (int c = t; c < 512; c += 256) {
            int r  = c >> 2;               // row / n: 0..127
            int cc = (c & 3) * 8;          // u16 offset within 32-wide chunk
            int row = row0 + r;
            if (row < N) {
                async_copy_b128(sX_base + (unsigned)(r * 40 + cc) * 2,
                                xin + (size_t)row * FDIM + kc * 32 + cc);
            } else {
                *(uint4*)&sXx[r][cc] = make_uint4(0u, 0u, 0u, 0u);
            }
            async_copy_b128(sWn_base + (unsigned)(r * 40 + cc) * 2,
                            WnT + (size_t)r * FDIM + kc * 32 + cc);
            async_copy_b128(sWs_base + (unsigned)(r * 40 + cc) * 2,
                            WsT + (size_t)r * FDIM + kc * 32 + cc);
        }
        // --- stage agg tile (fp32 -> bf16 with -inf -> 0 fix) ---
        {
            int rr = t >> 3;               // 0..31
            int kk = (t & 7) * 4;          // 0,4,...,28
            #pragma unroll
            for (int r = rr; r < 128; r += 32) {
                int row = row0 + r;
                float4 va = make_float4(0.f, 0.f, 0.f, 0.f);
                if (row < N)
                    va = *(const float4*)&agg[(long)row * FDIM + kc * 32 + kk];
                if (__float_as_uint(va.x) == 0xFF800000u) va.x = 0.f;
                if (__float_as_uint(va.y) == 0xFF800000u) va.y = 0.f;
                if (__float_as_uint(va.z) == 0xFF800000u) va.z = 0.f;
                if (__float_as_uint(va.w) == 0xFF800000u) va.w = 0.f;
                unsigned a0 = (unsigned)bf16b(va.x) | ((unsigned)bf16b(va.y) << 16);
                unsigned a1 = (unsigned)bf16b(va.z) | ((unsigned)bf16b(va.w) << 16);
                *(uint2*)&sAg[r][kk] = make_uint2(a0, a1);
            }
        }
        wait_async0();                     // this wave's async copies landed
        __syncthreads();                   // all waves' staging visible

        // --- A fragments (16x32 bf16): lanes 0-15 K 0-7/16-23, lanes 16-31 K 8-15/24-31
        const int m  = wv * 16 + (lane & 15);
        const int kg = lane >> 4;
        Frag16 aAg, aXx;
        aAg.q[0] = *(const uint4*)&sAg[m][kg * 8];
        aAg.q[1] = *(const uint4*)&sAg[m][kg * 8 + 16];
        aXx.q[0] = *(const uint4*)&sXx[m][kg * 8];
        aXx.q[1] = *(const uint4*)&sXx[m][kg * 8 + 16];

        // --- B fragments (32x16 bf16): n = lane&15, lanes 0-15 K 0-7/16-23, lanes 16-31 K 8-15/24-31
        const int bn = lane & 15;
        const int bk = (lane >> 4) * 8;
        #pragma unroll
        for (int nt = 0; nt < 8; ++nt) {
            Frag16 bN, bS;
            bN.q[0] = *(const uint4*)&sWn[nt * 16 + bn][bk];
            bN.q[1] = *(const uint4*)&sWn[nt * 16 + bn][bk + 16];
            bS.q[0] = *(const uint4*)&sWs[nt * 16 + bn][bk];
            bS.q[1] = *(const uint4*)&sWs[nt * 16 + bn][bk + 16];
            acc[nt] = __builtin_amdgcn_wmma_f32_16x16x32_bf16(
                false, aAg.v, false, bN.v, (short)0, acc[nt], false, false);
            acc[nt] = __builtin_amdgcn_wmma_f32_16x16x32_bf16(
                false, aXx.v, false, bS.v, (short)0, acc[nt], false, false);
        }
    }

    // --- epilogue: C/D layout lane l: n = l&15, m = r + (l>=16)*8 ---
    const int cn   = lane & 15;
    const int rsub = (lane >> 4) * 8;
    #pragma unroll
    for (int nt = 0; nt < 8; ++nt) {
        int col  = nt * 16 + cn;
        float bv = bias[col];
        #pragma unroll
        for (int r = 0; r < 8; ++r) {
            int row = row0 + wv * 16 + rsub + r;
            if (row < N) {
                float v = acc[nt][r] + bv;
                xout[(long)row * FDIM + col] = bf16b(v > 0.f ? v : 0.f);
            }
        }
    }
}

// one thread per (node, 8-feature chunk): scatter-max into pooled[batch[n]]
__global__ void pool_scatter_kernel(const unsigned short* __restrict__ xb,
                                    const long long* __restrict__ batch,
                                    float* __restrict__ pooled, int N) {
    long i = (long)blockIdx.x * blockDim.x + threadIdx.x;
    long n = i >> 4;
    int  c = (int)(i & 15) * 8;
    if (n >= N) return;
    long g = (long)batch[n];
    uint4 v = *(const uint4*)&xb[n * FDIM + c];
    float* p = &pooled[g * FDIM + c];
    atomicMaxF(p + 0, bf16f((unsigned short)(v.x & 0xFFFF)));
    atomicMaxF(p + 1, bf16f((unsigned short)(v.x >> 16)));
    atomicMaxF(p + 2, bf16f((unsigned short)(v.y & 0xFFFF)));
    atomicMaxF(p + 3, bf16f((unsigned short)(v.y >> 16)));
    atomicMaxF(p + 4, bf16f((unsigned short)(v.z & 0xFFFF)));
    atomicMaxF(p + 5, bf16f((unsigned short)(v.z >> 16)));
    atomicMaxF(p + 6, bf16f((unsigned short)(v.w & 0xFFFF)));
    atomicMaxF(p + 7, bf16f((unsigned short)(v.w >> 16)));
}

// h[g][j] = pooled[g] . lin_W[:,j] + lin_b[j]   (64 x 256, tiny)
__global__ void mlp1_kernel(const float* __restrict__ pooled,
                            const float* __restrict__ W,
                            const float* __restrict__ b,
                            float* __restrict__ h) {
    int i = blockIdx.x * blockDim.x + threadIdx.x;
    if (i >= NGRAPH * LINDIM) return;
    int g = i / LINDIM, j = i % LINDIM;
    float s = b[j];
    #pragma unroll 8
    for (int k = 0; k < FDIM; ++k)
        s += pooled[g * FDIM + k] * W[k * LINDIM + j];
    h[i] = s;
}

// out[g][j] = h[g] . out_W[:,j] + out_b[j]   (64 x 10, tiny)
__global__ void mlp2_kernel(const float* __restrict__ h,
                            const float* __restrict__ W,
                            const float* __restrict__ b,
                            float* __restrict__ out) {
    int i = blockIdx.x * blockDim.x + threadIdx.x;
    if (i >= NGRAPH * OUTDIM) return;
    int g = i / OUTDIM, j = i % OUTDIM;
    float s = b[j];
    #pragma unroll 8
    for (int k = 0; k < LINDIM; ++k)
        s += h[g * LINDIM + k] * W[k * OUTDIM + j];
    out[i] = s;
}

// ---------------- launcher ----------------

extern "C" void kernel_launch(void* const* d_in, const int* in_sizes, int n_in,
                              void* d_out, int out_size, void* d_ws, size_t ws_size,
                              hipStream_t stream) {
    const float*     x       = (const float*)d_in[0];
    const long long* ei      = (const long long*)d_in[1];   // int64 [2,E]
    const long long* batch   = (const long long*)d_in[2];   // int64 [N]
    const float*     Ws_self = (const float*)d_in[3];       // [3,128,128]
    const float*     Ws_nei  = (const float*)d_in[4];       // [3,128,128]
    const float*     bs      = (const float*)d_in[5];       // [3,128]
    const float*     lin_W   = (const float*)d_in[6];       // [128,256]
    const float*     lin_b   = (const float*)d_in[7];       // [256]
    const float*     out_W   = (const float*)d_in[8];       // [256,10]
    const float*     out_b   = (const float*)d_in[9];       // [10]
    float*           out     = (float*)d_out;

    const int  N = in_sizes[0] / FDIM;
    const long E = (long)in_sizes[1] / 2;
    const long long* src = ei;
    const long long* dst = ei + E;

    // workspace carve-up (bytes)
    char* w = (char*)d_ws;
    float* agg = (float*)w;                 w += (size_t)N * FDIM * 4;
    unsigned short* xb0 = (unsigned short*)w; w += (size_t)N * FDIM * 2;
    unsigned short* xbA = (unsigned short*)w; w += (size_t)N * FDIM * 2;
    unsigned short* xbB = (unsigned short*)w; w += (size_t)N * FDIM * 2;
    unsigned short* WTn[3]; unsigned short* WTs[3];
    for (int l = 0; l < 3; ++l) {
        WTn[l] = (unsigned short*)w; w += (size_t)FDIM * FDIM * 2;
        WTs[l] = (unsigned short*)w; w += (size_t)FDIM * FDIM * 2;
    }
    float* pooled = (float*)w;              w += (size_t)NGRAPH * FDIM * 4;
    float* h      = (float*)w;              w += (size_t)NGRAPH * LINDIM * 4;

    // --- prologue: x -> bf16, weights -> bf16 transposed [n][k] ---
    const long n2 = (long)N * FDIM / 2;
    cvt_f32_bf16_kernel<<<(int)((n2 + 255) / 256), 256, 0, stream>>>(x, xb0, n2);
    for (int l = 0; l < 3; ++l) {
        wtrans_kernel<<<(FDIM * FDIM + 255) / 256, 256, 0, stream>>>(
            Ws_nei + (size_t)l * FDIM * FDIM, WTn[l]);
        wtrans_kernel<<<(FDIM * FDIM + 255) / 256, 256, 0, stream>>>(
            Ws_self + (size_t)l * FDIM * FDIM, WTs[l]);
    }

    const long nAgg = (long)N * FDIM;
    const long nSc  = E * 16;
    const long nPo  = (long)N * 16;

    const unsigned short* xcur = xb0;
    unsigned short* bufs[2] = {xbA, xbB};
    for (int l = 0; l < 3; ++l) {
        fill_u32_kernel<<<(int)((nAgg + 1023) / 1024), 1024, 0, stream>>>(
            (unsigned*)agg, 0xFF800000u, nAgg);
        scatter_max_kernel<<<(int)((nSc + 255) / 256), 256, 0, stream>>>(
            xcur, src, dst, agg, E);
        unsigned short* xnext = bufs[l & 1];
        gnn_layer_kernel<<<(N + 127) / 128, 256, 0, stream>>>(
            agg, xcur, WTn[l], WTs[l], bs + (size_t)l * FDIM, xnext, N);
        xcur = xnext;
    }

    fill_u32_kernel<<<(NGRAPH * FDIM + 1023) / 1024, 1024, 0, stream>>>(
        (unsigned*)pooled, 0xFF800000u, NGRAPH * FDIM);
    pool_scatter_kernel<<<(int)((nPo + 255) / 256), 256, 0, stream>>>(
        xcur, batch, pooled, N);
    fix_neginf_kernel<<<(NGRAPH * FDIM + 255) / 256, 256, 0, stream>>>(
        pooled, NGRAPH * FDIM);
    mlp1_kernel<<<(NGRAPH * LINDIM + 255) / 256, 256, 0, stream>>>(
        pooled, lin_W, lin_b, h);
    mlp2_kernel<<<(NGRAPH * OUTDIM + 255) / 256, 256, 0, stream>>>(
        h, out_W, out_b, out);
}
